// GraphConvBlock_11940009083129
// MI455X (gfx1250) — matrix operand order repaired and checked by reading, not compiled, not run on hardware
//
#include <hip/hip_runtime.h>
#include <hip/hip_bf16.h>

#define N_NODES 50000
#define N_EDGES 800000
#define IN_DIM  128
#define OUT_DIM 256
#define N_TILES (N_NODES / 16)   // 3125 exactly

typedef __attribute__((ext_vector_type(16))) __bf16 v16bf;
typedef __attribute__((ext_vector_type(8)))  float  v8f;

union AFrag { v16bf v; unsigned u[8]; };
union BFrag { v16bf v; uint4 q[2];    };

__device__ __forceinline__ float fast_silu(float x) {
  // x * sigmoid(x) with native v_rcp_f32 (no IEEE div chain)
  return x * __builtin_amdgcn_rcpf(1.0f + __expf(-x));
}

// ---------------- workspace zero ----------------
__global__ void zero_kernel(float* __restrict__ p, int n) {
  int i = blockIdx.x * blockDim.x + threadIdx.x;
  for (; i < n; i += gridDim.x * blockDim.x) p[i] = 0.0f;
}

// ---------------- degree accumulation ----------------
__global__ void degree_kernel(const int* __restrict__ src, const int* __restrict__ dst,
                              float* __restrict__ deg_out, float* __restrict__ deg_in) {
  int e = blockIdx.x * blockDim.x + threadIdx.x;
  if (e >= N_EDGES) return;
  atomicAdd(&deg_out[src[e]], 1.0f);
  atomicAdd(&deg_in[dst[e]], 1.0f);
}

// ---------------- pack fp32 weights [K][256] into bf16 WMMA B-fragment order ----------
// Fragment (kt,nt) -> 32 lanes -> 16 bf16 per lane (contiguous, 32B/lane).
// B 32x16 bf16 layout: lane L holds column N = nt*16 + (L&15); element j maps to
// K = kt*32 + (L>>4)*16 + j.
__global__ void pack_w_kernel(const float* __restrict__ W, __bf16* __restrict__ P, int Ktot) {
  int pidx = blockIdx.x * blockDim.x + threadIdx.x;
  if (pidx >= Ktot * OUT_DIM) return;
  int j    = pidx & 15;
  int lane = (pidx >> 4) & 31;
  int frag = pidx >> 9;
  int nt   = frag & 15;
  int kt   = frag >> 4;
  int n = nt * 16 + (lane & 15);
  int k = kt * 32 + (lane >> 4) * 16 + j;
  P[pidx] = (__bf16)W[k * OUT_DIM + n];
}

// ---------------- edge gather/scatter: agg[dst] += feats[src] * rsqrt(deg_out[src]) ----
// One wave per edge; edge indices / degree are wave-uniform -> force scalar loads.
__global__ __launch_bounds__(256) void scatter_kernel(
    const float* __restrict__ feats, const float* __restrict__ deg_out,
    const int* __restrict__ src, const int* __restrict__ dst,
    float* __restrict__ agg) {
  int e = __builtin_amdgcn_readfirstlane(blockIdx.x * 8 + (threadIdx.x >> 5));
  if (e >= N_EDGES) return;
  int c = threadIdx.x & 31;                               // 32 float4 chunks per row
  int s = __builtin_amdgcn_readfirstlane(src[e]);
  int d = __builtin_amdgcn_readfirstlane(dst[e]);
  float sc = rsqrtf(fmaxf(deg_out[s], 1.0f));
  float4 f = reinterpret_cast<const float4*>(feats)[s * (IN_DIM / 4) + c];
  float* ap = agg + d * IN_DIM + c * 4;
  atomicAdd(ap + 0, f.x * sc);
  atomicAdd(ap + 1, f.y * sc);
  atomicAdd(ap + 2, f.z * sc);
  atomicAdd(ap + 3, f.w * sc);
}

// ---------------- one dense layer: h(16xK bf16 in LDS) @ W(packed) + b -> silu ---------
template<int KTILES, bool TO_GLOBAL>
__device__ __forceinline__ void run_layer(__bf16* h, const unsigned* __restrict__ Wp,
                                          const float* __restrict__ bias,
                                          float* __restrict__ out, int tile, int lane) {
  const int half = lane >> 4;
  const int col  = lane & 15;
  // --- hoist all A fragments into registers (16-bit A 16x32, contiguous k = 16*half + element) ---
  AFrag a[KTILES];
  const unsigned* hrow = reinterpret_cast<const unsigned*>(h) + col * (OUT_DIM / 2);
  #pragma unroll
  for (int kt = 0; kt < KTILES; ++kt) {
    #pragma unroll
    for (int v = 0; v < 8; ++v) {
      int kp = (v << 1) + (half << 4);
      a[kt].u[v] = hrow[(kt * 32 + kp) >> 1];   // aligned ds_load_b32 (K-pair)
    }
  }
  // Process 2 n-tiles per iteration: two independent accumulator chains so the
  // scheduler can interleave WMMAs and hide the post-WMMA hazard NOPs.
  for (int nt = 0; nt < 16; nt += 2) {
    float bv0 = bias[nt * 16 + col];
    float bv1 = bias[(nt + 1) * 16 + col];
    v8f acc0 = {bv0, bv0, bv0, bv0, bv0, bv0, bv0, bv0};
    v8f acc1 = {bv1, bv1, bv1, bv1, bv1, bv1, bv1, bv1};
    #pragma unroll
    for (int kt = 0; kt < KTILES; ++kt) {
      BFrag b0, b1;
      const uint4* bp0 =
          reinterpret_cast<const uint4*>(Wp + ((kt * 16 + nt) * 32 + lane) * 8);
      const uint4* bp1 =
          reinterpret_cast<const uint4*>(Wp + ((kt * 16 + nt + 1) * 32 + lane) * 8);
      b0.q[0] = bp0[0]; b0.q[1] = bp0[1];
      b1.q[0] = bp1[0]; b1.q[1] = bp1[1];
      acc0 = __builtin_amdgcn_wmma_f32_16x16x32_bf16(
          false, a[kt].v, false, b0.v, (short)0, acc0, false, false);
      acc1 = __builtin_amdgcn_wmma_f32_16x16x32_bf16(
          false, a[kt].v, false, b1.v, (short)0, acc1, false, false);
    }
    #pragma unroll
    for (int r = 0; r < 8; ++r) {
      acc0[r] = fast_silu(acc0[r]);
      acc1[r] = fast_silu(acc1[r]);
    }
    if (TO_GLOBAL) {
      #pragma unroll
      for (int r = 0; r < 8; ++r) {             // C/D layout: M = r + 8*half, N = col
        out[(tile * 16 + r + half * 8) * OUT_DIM + nt * 16 + col]       = acc0[r];
        out[(tile * 16 + r + half * 8) * OUT_DIM + (nt + 1) * 16 + col] = acc1[r];
      }
    } else {
      #pragma unroll
      for (int r = 0; r < 8; ++r) {
        h[(r + half * 8) * OUT_DIM + nt * 16 + col]       = (__bf16)acc0[r];
        h[(r + half * 8) * OUT_DIM + (nt + 1) * 16 + col] = (__bf16)acc1[r];
      }
    }
  }
}

// ---------------- fused 3-layer MLP; one 16-node tile per wave, no barriers -----------
__global__ __launch_bounds__(256) void mlp_kernel(
    const float* __restrict__ agg, const float* __restrict__ deg_in,
    const unsigned* __restrict__ pWc, const unsigned* __restrict__ pW1,
    const unsigned* __restrict__ pW2,
    const float* __restrict__ bC, const float* __restrict__ bF1,
    const float* __restrict__ bF2, float* __restrict__ out) {
  __shared__ __bf16 lds[8 * 16 * OUT_DIM];      // 64 KB: 8 KB per wave
  const int lane = threadIdx.x & 31;
  const int wave = __builtin_amdgcn_readfirstlane(threadIdx.x >> 5);
  const int tile = blockIdx.x * 8 + wave;
  if (tile >= N_TILES) return;
  __bf16* h = &lds[wave * 16 * OUT_DIM];

  // stage x = agg * rsqrt(deg_in) as bf16, row-major [16][128] (stride 256)
  const float4* agg4 = reinterpret_cast<const float4*>(agg);
  #pragma unroll
  for (int row = 0; row < 16; ++row) {
    int node = tile * 16 + row;
    float s = rsqrtf(fmaxf(deg_in[node], 1.0f));
    float4 f = agg4[node * (IN_DIM / 4) + lane];
    __bf16* p = &h[row * OUT_DIM + lane * 4];
    p[0] = (__bf16)(f.x * s);
    p[1] = (__bf16)(f.y * s);
    p[2] = (__bf16)(f.z * s);
    p[3] = (__bf16)(f.w * s);
  }

  run_layer<4, false>(h, pWc, bC,  out, tile, lane);   // 128 -> 256, silu
  run_layer<8, false>(h, pW1, bF1, out, tile, lane);   // 256 -> 256, silu
  run_layer<8, true >(h, pW2, bF2, out, tile, lane);   // 256 -> 256, silu -> global
}

extern "C" void kernel_launch(void* const* d_in, const int* in_sizes, int n_in,
                              void* d_out, int out_size, void* d_ws, size_t ws_size,
                              hipStream_t stream) {
  const float* feats = (const float*)d_in[0];
  const float* Wc    = (const float*)d_in[1];
  const float* bc    = (const float*)d_in[2];
  const float* W1    = (const float*)d_in[3];
  const float* b1    = (const float*)d_in[4];
  const float* W2    = (const float*)d_in[5];
  const float* b2    = (const float*)d_in[6];
  const int*   src   = (const int*)d_in[7];
  const int*   dst   = (const int*)d_in[8];
  float* out = (float*)d_out;

  // workspace layout (all 16B aligned)
  float*  deg_out = (float*)d_ws;                           // 50000 f32
  float*  deg_in  = deg_out + N_NODES;                      // 50000 f32
  float*  agg     = deg_in + N_NODES;                       // 6.4M f32
  __bf16* pWc     = (__bf16*)(agg + (size_t)N_NODES * IN_DIM);
  __bf16* pW1     = pWc + IN_DIM * OUT_DIM;
  __bf16* pW2     = pW1 + OUT_DIM * OUT_DIM;

  const int zn = 2 * N_NODES + N_NODES * IN_DIM;
  zero_kernel<<<4096, 256, 0, stream>>>(deg_out, zn);
  degree_kernel<<<(N_EDGES + 255) / 256, 256, 0, stream>>>(src, dst, deg_out, deg_in);
  pack_w_kernel<<<(IN_DIM * OUT_DIM + 255) / 256, 256, 0, stream>>>(Wc, pWc, IN_DIM);
  pack_w_kernel<<<(OUT_DIM * OUT_DIM + 255) / 256, 256, 0, stream>>>(W1, pW1, OUT_DIM);
  pack_w_kernel<<<(OUT_DIM * OUT_DIM + 255) / 256, 256, 0, stream>>>(W2, pW2, OUT_DIM);
  scatter_kernel<<<(N_EDGES + 7) / 8, 256, 0, stream>>>(feats, deg_out, src, dst, agg);
  mlp_kernel<<<(N_TILES + 7) / 8, 256, 0, stream>>>(
      agg, deg_in, (const unsigned*)pWc, (const unsigned*)pW1, (const unsigned*)pW2,
      bc, b1, b2, out);
}
